// GrowingCrystalAttention_69810398429662
// MI455X (gfx1250) — matrix operand order repaired and checked by reading, not compiled, not run on hardware
//
#include <hip/hip_runtime.h>
#include <hip/hip_bf16.h>
#include <stdint.h>

// ---------------------------------------------------------------------------
// GrowingCrystalAttention on MI455X (gfx1250, wave32, WMMA)
//   B=4, T=512, D=512, N=128  -> BT = 2048 tokens
//   einsum('btn,btd,nde->bte') == GEMM (2048 x 65536) x (65536 x 512), A formed
//   on the fly as attn[bt,n] * x[bt,d].  v_wmma_f32_16x16x32_bf16, fp32 acc,
//   double-buffered LDS, 128x64 tile / 8 waves.  value_weight is pre-transposed
//   + converted once to bf16 [e][k] so the hot loop's B path is a pure copy.
// ---------------------------------------------------------------------------

typedef __attribute__((ext_vector_type(16))) __bf16 v16bf;
typedef __attribute__((ext_vector_type(8)))  __bf16 v8bf;
typedef __attribute__((ext_vector_type(4)))  __bf16 v4bf;
typedef __attribute__((ext_vector_type(2)))  __bf16 v2bf;
typedef __attribute__((ext_vector_type(8)))  float  v8f;

#define BT_TOT   2048   // B*T
#define DDIM     512
#define NNEUR    128
#define KTOT     (NNEUR * DDIM)   // 65536

#define BM       128    // output rows per workgroup
#define BN       64     // output cols per workgroup
#define BK       32     // K slice per iteration (one WMMA K step)
#define THREADS  256    // 8 waves of 32
#define A_STRIDE 40     // bf16 elems per LDS row (32 + 8 pad)
#define B_STRIDE 40

// B-source flavors for the unified GEMM
#define MODE_EINSUM_BF16 0   // B = vw_bf16 [e][K], bf16, K-contiguous
#define MODE_LINEAR      1   // B = out_W   [e][K], fp32, K-contiguous (+bias)
#define MODE_EINSUM_F32  2   // B = value_weight [k][e], fp32 (fallback: transpose on the fly)

__device__ __forceinline__ __bf16 cvt_bf16(float f) { return (__bf16)f; }

// ---------------------------------------------------------------------------
// Kernel 1: attn[bt,n] = softmax_n( scales[n] / (||x[bt]-pos[n]|| + 0.1) )
// ---------------------------------------------------------------------------
__global__ __launch_bounds__(128)
void crystal_attn_kernel(const float* __restrict__ x,
                         const float* __restrict__ pos,
                         const float* __restrict__ scales,
                         float* __restrict__ attn) {
    __shared__ float4 xs[DDIM / 4];
    __shared__ float  red[NNEUR];

    const int bt  = blockIdx.x;
    const int tid = threadIdx.x;

    xs[tid] = ((const float4*)x)[(size_t)bt * (DDIM / 4) + tid];
    __syncthreads();

    const float4* p4 = (const float4*)pos + (size_t)tid * (DDIM / 4);
    float d2 = 0.f;
#pragma unroll 4
    for (int j = 0; j < DDIM / 4; ++j) {
        float4 p  = p4[j];
        float4 xv = xs[j];
        float dx = xv.x - p.x, dy = xv.y - p.y, dz = xv.z - p.z, dw = xv.w - p.w;
        d2 += dx * dx + dy * dy + dz * dz + dw * dw;
    }
    float val = scales[tid] / (sqrtf(d2) + 0.1f);

    red[tid] = val;
    __syncthreads();
    for (int s = NNEUR / 2; s > 0; s >>= 1) {
        if (tid < s) red[tid] = fmaxf(red[tid], red[tid + s]);
        __syncthreads();
    }
    float m = red[0];
    __syncthreads();

    float e = __expf(val - m);
    red[tid] = e;
    __syncthreads();
    for (int s = NNEUR / 2; s > 0; s >>= 1) {
        if (tid < s) red[tid] += red[tid + s];
        __syncthreads();
    }
    attn[(size_t)bt * NNEUR + tid] = e / red[0];
}

// ---------------------------------------------------------------------------
// Kernel 2: one-time transpose+convert  vw_bf16[e][k] = bf16(value_weight[k][e])
// 64x64 tiles via LDS so both global sides stay coalesced.
// ---------------------------------------------------------------------------
#define T_STRIDE 68
__global__ __launch_bounds__(256)
void vw_transpose_kernel(const float* __restrict__ vw,   // [KTOT][DDIM] fp32
                         __bf16* __restrict__ vwb) {     // [DDIM][KTOT] bf16
    __shared__ __bf16 t[64 * T_STRIDE];                  // t[e][k] within tile

    const int k0  = blockIdx.x * 64;
    const int e0  = blockIdx.y * 64;
    const int tid = threadIdx.x;
    const int c4  = (tid & 15) * 4;     // 0..60
    const int r   = tid >> 4;           // 0..15

#pragma unroll
    for (int i = 0; i < 4; ++i) {
        const int kr = r + i * 16;      // k row within tile
        float4 v = *(const float4*)(vw + (size_t)(k0 + kr) * DDIM + e0 + c4);
        t[(c4 + 0) * T_STRIDE + kr] = cvt_bf16(v.x);
        t[(c4 + 1) * T_STRIDE + kr] = cvt_bf16(v.y);
        t[(c4 + 2) * T_STRIDE + kr] = cvt_bf16(v.z);
        t[(c4 + 3) * T_STRIDE + kr] = cvt_bf16(v.w);
    }
    __syncthreads();
#pragma unroll
    for (int i = 0; i < 4; ++i) {
        const int er = r + i * 16;      // e row within tile
        v4bf pk = *(const v4bf*)&t[er * T_STRIDE + c4];
        *(v4bf*)(vwb + (size_t)(e0 + er) * KTOT + k0 + c4) = pk;
    }
}

// ---------------------------------------------------------------------------
// Tile loader: global -> (scale) -> bf16 -> LDS for the K-step at kk.
// ---------------------------------------------------------------------------
__device__ __forceinline__ void load_tile(const float* __restrict__ Ax,
                                          const float* __restrict__ attn,
                                          const void* __restrict__ Bsrc,
                                          int mode, int kk, int K,
                                          int rowBase, int colBase, int tid,
                                          __bf16* __restrict__ As,
                                          __bf16* __restrict__ Bs) {
    const int n  = kk >> 9;
    const int d0 = kk & (DDIM - 1);

    // ---- A tile: 128 rows x 32 k, 2 threads per row (16 floats each) ----
    {
        const int r  = tid >> 1;
        const int h  = tid & 1;
        const int bt = rowBase + r;
        const float* src = Ax + (size_t)bt * DDIM + d0 + h * 16;
        float scale = 1.0f;
        if (mode != MODE_LINEAR) scale = attn[(size_t)bt * NNEUR + n];
        float f[16];
#pragma unroll
        for (int q = 0; q < 4; ++q) {
            float4 v = ((const float4*)src)[q];
            f[q * 4 + 0] = v.x * scale;
            f[q * 4 + 1] = v.y * scale;
            f[q * 4 + 2] = v.z * scale;
            f[q * 4 + 3] = v.w * scale;
        }
        v8bf lo, hi;
#pragma unroll
        for (int q = 0; q < 8; ++q) { lo[q] = cvt_bf16(f[q]); hi[q] = cvt_bf16(f[8 + q]); }
        *(v8bf*)&As[r * A_STRIDE + h * 16 + 0] = lo;
        *(v8bf*)&As[r * A_STRIDE + h * 16 + 8] = hi;
    }

    // ---- B tile: 32 k x 64 e, stored K-contiguous per column ----
    if (mode == MODE_EINSUM_BF16) {
        // pre-transposed bf16: pure copy, zero VALU in the hot path
        const __bf16* Bv = (const __bf16*)Bsrc;
        const int e  = tid >> 2;               // 0..63
        const int qh = tid & 3;                // 8 bf16 each
        const __bf16* src = Bv + (size_t)(colBase + e) * KTOT + kk + qh * 8;
        *(v8bf*)&Bs[e * B_STRIDE + qh * 8] = *(const v8bf*)src;
        if (kk + 2 * BK < K)
            __builtin_prefetch(Bv + (size_t)(colBase + e) * KTOT + kk + 2 * BK + qh * 8, 0, 1);
    } else if (mode == MODE_LINEAR) {
        const float* Bv = (const float*)Bsrc;  // out_W [e][k] fp32
        const int e  = tid >> 2;
        const int qh = tid & 3;
        const float* src = Bv + (size_t)(colBase + e) * DDIM + kk + qh * 8;
        v8bf pk;
#pragma unroll
        for (int q = 0; q < 8; ++q) pk[q] = cvt_bf16(src[q]);
        *(v8bf*)&Bs[e * B_STRIDE + qh * 8] = pk;
    } else {
        // fallback: fp32 K-major value_weight, transpose on the fly
        const float* Bv = (const float*)Bsrc;
        const int e4 = (tid & 15) * 4;
        const int k2 = (tid >> 4) * 2;
        float f[2][4];
#pragma unroll
        for (int i = 0; i < 2; ++i) {
            float4 v = *(const float4*)(Bv + (size_t)(kk + k2 + i) * DDIM + colBase + e4);
            f[i][0] = v.x; f[i][1] = v.y; f[i][2] = v.z; f[i][3] = v.w;
        }
#pragma unroll
        for (int j = 0; j < 4; ++j) {
            v2bf pk;
            pk[0] = cvt_bf16(f[0][j]);
            pk[1] = cvt_bf16(f[1][j]);
            *(v2bf*)&Bs[(e4 + j) * B_STRIDE + k2] = pk;
        }
        if (kk + 2 * BK < K)
            __builtin_prefetch(Bv + (size_t)(kk + 2 * BK + k2) * DDIM + colBase + e4, 0, 1);
    }
}

// ---------------------------------------------------------------------------
// Kernel 3: C(2048 x 512) = A(2048 x K) * B(K x 512), bf16 WMMA, fp32 acc
// 128x64 tile / WG; 8 waves; each wave: 32 rows x 64 cols = 8 WMMA tiles.
// Double-buffered LDS: one barrier per K-step, loads overlap WMMA.
// ---------------------------------------------------------------------------
__global__ __launch_bounds__(THREADS)
void crystal_gemm_kernel(const float* __restrict__ Ax,
                         const float* __restrict__ attn,
                         const void* __restrict__ Bsrc,
                         const float* __restrict__ bias,
                         float* __restrict__ Cout,
                         int K, int mode) {
    __shared__ __bf16 Asm[2][BM * A_STRIDE];
    __shared__ __bf16 Bsm[2][BN * B_STRIDE];

    const int tid     = threadIdx.x;
    const int wave    = tid >> 5;
    const int lane    = tid & 31;
    const int half    = lane >> 4;
    const int lrow    = lane & 15;
    const int rowBase = blockIdx.y * BM;
    const int colBase = blockIdx.x * BN;

    v8f acc[2][4];
#pragma unroll
    for (int rt = 0; rt < 2; ++rt)
#pragma unroll
        for (int ct = 0; ct < 4; ++ct)
#pragma unroll
            for (int i = 0; i < 8; ++i) acc[rt][ct][i] = 0.f;

    const int nIter = K / BK;
    load_tile(Ax, attn, Bsrc, mode, 0, K, rowBase, colBase, tid, Asm[0], Bsm[0]);

    for (int iter = 0; iter < nIter; ++iter) {
        const int cur = iter & 1;
        __syncthreads();   // buf[cur] ready; buf[cur^1] fully consumed

        if (iter + 1 < nIter)
            load_tile(Ax, attn, Bsrc, mode, (iter + 1) * BK, K,
                      rowBase, colBase, tid, Asm[cur ^ 1], Bsm[cur ^ 1]);

        const __bf16* As = Asm[cur];
        const __bf16* Bs = Bsm[cur];

        // 16-bit A 16x32 layout: lanes 0-15 hold K {0..7,16..23}, lanes 16-31 K {8..15,24..31}
        v16bf afrag[2];
#pragma unroll
        for (int rt = 0; rt < 2; ++rt) {
            const int aRow = wave * 32 + rt * 16 + lrow;
            v8bf alo = *(const v8bf*)&As[aRow * A_STRIDE + half * 8];
            v8bf ahi = *(const v8bf*)&As[aRow * A_STRIDE + 16 + half * 8];
            afrag[rt] = __builtin_shufflevector(alo, ahi,
                            0, 1, 2, 3, 4, 5, 6, 7, 8, 9, 10, 11, 12, 13, 14, 15);
        }
#pragma unroll
        for (int ct = 0; ct < 4; ++ct) {
            // 16-bit B 32x16: lanes 0-15 hold K {0..7,16..23}, lanes 16-31 K {8..15,24..31}, col N=lane&15
            const int bCol = ct * 16 + lrow;
            v8bf blo = *(const v8bf*)&Bs[bCol * B_STRIDE + half * 8];
            v8bf bhi = *(const v8bf*)&Bs[bCol * B_STRIDE + 16 + half * 8];
            v16bf bfrag = __builtin_shufflevector(blo, bhi,
                              0, 1, 2, 3, 4, 5, 6, 7, 8, 9, 10, 11, 12, 13, 14, 15);
#pragma unroll
            for (int rt = 0; rt < 2; ++rt)
                acc[rt][ct] = __builtin_amdgcn_wmma_f32_16x16x32_bf16(
                                  false, afrag[rt], false, bfrag,
                                  (short)0, acc[rt][ct], false, false);
        }
    }

    // ---- epilogue: C layout VGPR i -> M = i (+8 for upper lane half), N = lane&15
#pragma unroll
    for (int rt = 0; rt < 2; ++rt) {
        const int cRow0 = rowBase + wave * 32 + rt * 16 + ((lane < 16) ? 0 : 8);
#pragma unroll
        for (int ct = 0; ct < 4; ++ct) {
            const int cCol = colBase + ct * 16 + lrow;
            const float b = (mode == MODE_LINEAR) ? bias[cCol] : 0.f;
#pragma unroll
            for (int i = 0; i < 8; ++i)
                Cout[(size_t)(cRow0 + i) * DDIM + cCol] = acc[rt][ct][i] + b;
        }
    }
}

// ---------------------------------------------------------------------------
// launch
// ---------------------------------------------------------------------------
extern "C" void kernel_launch(void* const* d_in, const int* in_sizes, int n_in,
                              void* d_out, int out_size, void* d_ws, size_t ws_size,
                              hipStream_t stream) {
    (void)in_sizes; (void)n_in; (void)out_size;

    const float* x            = (const float*)d_in[0];   // (B,T,D)
    const float* positions    = (const float*)d_in[1];   // (N,D)
    const float* scales       = (const float*)d_in[2];   // (N,)
    const float* value_weight = (const float*)d_in[3];   // (N,D,D) == (K x E)
    const float* out_W        = (const float*)d_in[4];   // (D,D), row e, col k
    const float* out_b        = (const float*)d_in[5];   // (D,)
    float*       out          = (float*)d_out;           // (B,T,D)

    const size_t attnB = (size_t)BT_TOT * NNEUR * 4;     // 1 MB
    const size_t tmpB  = (size_t)BT_TOT * DDIM * 4;      // 4 MB
    const size_t vwbB  = (size_t)KTOT * DDIM * 2;        // 64 MiB

    float*  attn = (float*)d_ws;
    float*  tmp  = (float*)((char*)d_ws + attnB);
    __bf16* vwb  = (__bf16*)((char*)d_ws + attnB + tmpB);

    // 1) softmax attention weights
    crystal_attn_kernel<<<BT_TOT, 128, 0, stream>>>(x, positions, scales, attn);

    dim3 grid(DDIM / BN, BT_TOT / BM);   // (8, 16)

    // 2) einsum as fused WMMA GEMM, K = N*D = 65536
    if (ws_size >= attnB + tmpB + vwbB) {
        // one-time transpose+convert of value_weight -> bf16 [e][k]
        vw_transpose_kernel<<<dim3(KTOT / 64, DDIM / 64), 256, 0, stream>>>(value_weight, vwb);
        crystal_gemm_kernel<<<grid, THREADS, 0, stream>>>(
            x, attn, (const void*)vwb, nullptr, tmp, KTOT, MODE_EINSUM_BF16);
    } else {
        crystal_gemm_kernel<<<grid, THREADS, 0, stream>>>(
            x, attn, (const void*)value_weight, nullptr, tmp, KTOT, MODE_EINSUM_F32);
    }

    // 3) out = tmp @ out_W^T + out_b, K = 512
    crystal_gemm_kernel<<<grid, THREADS, 0, stream>>>(
        tmp, nullptr, (const void*)out_W, out_b, out, DDIM, MODE_LINEAR);
}
